// EquivariantInteractionPPBlock_62457414418909
// MI455X (gfx1250) — matrix operand order repaired and checked by reading, not compiled, not run on hardware
//
#include <hip/hip_runtime.h>

// ---------------------------------------------------------------------------
// DimeNet++ interaction block for MI455X (gfx1250).
// GEMMs: bf16 WMMA (v_wmma_f32_16x16x32_bf16), f32 accumulate.
// Weight staging: Tensor Data Mover (tensor_load_to_lds + s_wait_tensorcnt),
//   copying a pre-transposed bf16 weight image straight into LDS (240KB/16KB
//   rows in ONE descriptor) -- no per-thread load/cvt/ds_store prologue.
// Scatter/segment-sum: fp32 VALU + global_atomic_add_f32.
// silu via v_exp_f32/v_rcp_f32 hardware transcendentals.
// ---------------------------------------------------------------------------

#define E_EDGES   262144
#define T_TRIP    2097152
#define EMB       128
#define INTD      64
#define NRBF      6
#define NSBF      42

typedef __bf16 bf16_t;
typedef __attribute__((ext_vector_type(16))) __bf16 bf16x16;
typedef __attribute__((ext_vector_type(8)))  __bf16 bf16x8;
typedef __attribute__((ext_vector_type(8)))  float  f32x8;
typedef __attribute__((ext_vector_type(4)))  unsigned int u32x4;
typedef __attribute__((ext_vector_type(8)))  int i32x8;
typedef __attribute__((ext_vector_type(4)))  int i32x4;

#ifndef __has_builtin
#define __has_builtin(x) 0
#endif
#if __has_builtin(__builtin_amdgcn_tensor_load_to_lds) && \
    __has_builtin(__builtin_amdgcn_s_wait_tensorcnt)
#define HAVE_TDM 1
#else
#define HAVE_TDM 0
#endif

union FragU { bf16x16 v; bf16x8 h[2]; };

// silu(x) = x / (1 + e^-x), via hardware exp2 + rcp (v_exp_f32, v_rcp_f32).
__device__ __forceinline__ float silu_f(float x) {
  float e = __builtin_amdgcn_exp2f(-1.44269504088896f * x);
  return x * __builtin_amdgcn_rcpf(1.0f + e);
}

#if HAVE_TDM
// One TDM descriptor copies nrows x 16KB of bf16 weight image (contiguous in
// global, contiguous in LDS). D# layout per CDNA5 ISA:
//   group0: [1:0]=count=1, [63:32]=lds_addr, [120:64]=global_addr, type=2
//   group1: data_size=2B, tensor_dim0=tile_dim0=8192 elems,
//           tensor_dim1=tile_dim1=nrows, tensor_dim0_stride=8192
__device__ __forceinline__ void tdm_load_weights(const bf16_t* gsrc,
                                                 void* ldsDst, int nrows) {
  unsigned long long ga = (unsigned long long)gsrc;
  unsigned lds_addr = (unsigned)(unsigned long long)ldsDst;  // flat[31:0]=LDS
  const unsigned dim0 = 8192;  // elements per row (16KB of bf16)
  u32x4 g0;
  g0[0] = 1u;                   // count=1, is_restore=0, gather_mode=0
  g0[1] = lds_addr;
  g0[2] = (unsigned)ga;                                   // global_addr[31:0]
  g0[3] = (unsigned)((ga >> 32) & 0x01FFFFFFu) | (2u << 30);  // [56:32],type=2
  i32x8 g1;
  g1[0] = 0x10000;                                  // data_size=1 (2 bytes)
  g1[1] = (int)((dim0 & 0xFFFFu) << 16);            // tensor_dim0[15:0]
  g1[2] = (int)((dim0 >> 16) | ((unsigned)nrows << 16));  // td0 hi | td1 lo
  g1[3] = (int)((dim0 & 0xFFFFu) << 16);            // td1 hi(0) | tile_dim0
  g1[4] = nrows;                                    // tile_dim1 (tile_dim2=0)
  g1[5] = (int)dim0;                                // tensor_dim0_stride lo
  g1[6] = 0;
  g1[7] = 0;
  i32x4 z4 = {0, 0, 0, 0};
#if __clang_major__ >= 23
  i32x8 z8 = {0, 0, 0, 0, 0, 0, 0, 0};
  __builtin_amdgcn_tensor_load_to_lds(g0, g1, z4, z4, z8, 0);
#else
  __builtin_amdgcn_tensor_load_to_lds(g0, g1, z4, z4, 0);
#endif
}
#endif

// A fragment (16x32 bf16): lane L -> row M = L%16; half = L/16.
// VGPR v0..3 hold K = 8*half + (0..7); v4..7 hold K = 16 + 8*half + (0..7).
__device__ __forceinline__ bf16x16 ld_frag_A(const bf16_t* act, int ld, int mrow,
                                             int kt, int half) {
  FragU f;
  const bf16_t* p = act + mrow * ld + kt * 32 + half * 8;
  f.h[0] = *(const bf16x8*)(p);
  f.h[1] = *(const bf16x8*)(p + 16);
  return f.v;
}

// B fragment (32x16 bf16): lane L -> col N = L%16; half = L/16.
// VGPR v0..3 hold K = 8*half + (0..7); v4..7 hold K = 16 + 8*half + (0..7).
// Weights stored TRANSPOSED (wT[n*ldK + k]) so this is two b128s.
__device__ __forceinline__ bf16x16 ld_frag_B(const bf16_t* wT, int ldK, int ncol,
                                             int kt, int half) {
  FragU f;
  const bf16_t* p = wT + ncol * ldK + kt * 32 + half * 8;
  f.h[0] = *(const bf16x8*)(p);
  f.h[1] = *(const bf16x8*)(p + 16);
  return f.v;
}

__device__ __forceinline__ f32x8 wmma_bf16(bf16x16 a, bf16x16 b, f32x8 c) {
  return __builtin_amdgcn_wmma_f32_16x16x32_bf16(false, a, false, b,
                                                 (short)0, c, false, false);
}

// One [16 x 128] x [128 x 128] GEMM for a wave; post(nt, acc) per 16x16 tile.
template <typename Post>
__device__ __forceinline__ void square_gemm(const bf16_t* actIn, const bf16_t* wTw,
                                            const float* bias, int wave, int half,
                                            int nn, Post post) {
  bf16x16 a[4];
  int mrow = wave * 16 + nn;
#pragma unroll
  for (int kt = 0; kt < 4; ++kt) a[kt] = ld_frag_A(actIn, EMB, mrow, kt, half);
#pragma unroll
  for (int nt = 0; nt < 8; ++nt) {
    bf16x16 bf[4];
#pragma unroll
    for (int kt = 0; kt < 4; ++kt)
      bf[kt] = ld_frag_B(wTw, EMB, nt * 16 + nn, kt, half);
    f32x8 acc;
    float b = bias[nt * 16 + nn];
#pragma unroll
    for (int e = 0; e < 8; ++e) acc[e] = b;
#pragma unroll
    for (int kt = 0; kt < 4; ++kt) acc = wmma_bf16(a[kt], bf[kt], acc);
    post(nt, acc);
  }
}

// ---------------------------------------------------------------------------
// Kernel 1: collapse basis projections.
// ---------------------------------------------------------------------------
__global__ void dpp_precompute_basis(const float* __restrict__ w_rbf1,
                                     const float* __restrict__ w_rbf2,
                                     const float* __restrict__ w_sbf1,
                                     const float* __restrict__ w_sbf2,
                                     float* __restrict__ w_rbf,
                                     float* __restrict__ w_sbf) {
  int tid = threadIdx.x;
  for (int i = tid; i < NRBF * EMB; i += 256) {
    int j = i / EMB, c = i % EMB;
    float s = 0.f;
#pragma unroll
    for (int p = 0; p < 8; ++p) s += w_rbf1[j * 8 + p] * w_rbf2[p * EMB + c];
    w_rbf[i] = s;
  }
  for (int i = tid; i < NSBF * INTD; i += 256) {
    int j = i / INTD, c = i % INTD;
    float s = 0.f;
#pragma unroll
    for (int p = 0; p < 8; ++p) s += w_sbf1[j * 8 + p] * w_sbf2[p * INTD + c];
    w_sbf[i] = s;
  }
}

__global__ void dpp_zero_f32(float* __restrict__ p, long long n) {
  long long i = (long long)blockIdx.x * 256 + threadIdx.x;
  if (i < n) p[i] = 0.0f;
}

// ---------------------------------------------------------------------------
// Kernel 2: build pre-transposed bf16 weight images for TDM staging.
// Tail image (122880 elems): [wTup 128n x 64k][7 x wT 128n x 128k]
// Head image  (40960 elems): [wTji][wTkj][wTdn 64n x 128k]
// ---------------------------------------------------------------------------
__global__ void dpp_prep_weights(
    const float* __restrict__ w_up, const float* __restrict__ wb1,
    const float* __restrict__ wb2, const float* __restrict__ w_fin,
    const float* __restrict__ wa1, const float* __restrict__ wa2,
    const float* __restrict__ w_ji, const float* __restrict__ w_kj,
    const float* __restrict__ w_down,
    bf16_t* __restrict__ wtail, bf16_t* __restrict__ whead) {
  long long i = (long long)blockIdx.x * 256 + threadIdx.x;
  if (i < 8192) {                       // wTup[n*64+k] = w_up[k*128+n]
    int n = (int)(i >> 6), k = (int)(i & 63);
    wtail[i] = (bf16_t)w_up[k * 128 + n];
  } else if (i < 122880) {
    long long j = i - 8192;
    int w = (int)(j >> 14);
    int r = (int)(j & 16383);
    int n = r >> 7, k = r & 127;
    const float* W[7] = {wb1, wb2, w_fin, wa1, wa2,
                         wa1 + EMB * EMB, wa2 + EMB * EMB};
    wtail[i] = (bf16_t)W[w][k * 128 + n];
  }
  if (i < 16384) {
    int n = (int)(i >> 7), k = (int)(i & 127);
    whead[i] = (bf16_t)w_ji[k * 128 + n];
  } else if (i < 32768) {
    long long j = i - 16384;
    int n = (int)(j >> 7), k = (int)(j & 127);
    whead[i] = (bf16_t)w_kj[k * 128 + n];
  } else if (i < 40960) {
    long long j = i - 32768;
    int n = (int)(j >> 7), k = (int)(j & 127);
    whead[i] = (bf16_t)w_down[k * 64 + n];
  }
}

// ---------------------------------------------------------------------------
// Kernel 3: fused head. Per 128-edge tile:
//   x_ji      = silu(x0@w_ji + b_ji)                 -> global f32 [E,128]
//   x_kj      = silu(x0@w_kj + b_kj) * rbf_e
//   x_kj_down = silu(x_kj @ w_down)                  -> global f32 [E,64]
// Weights arrive via one TDM descriptor (5 x 16KB rows into smem+0).
// ---------------------------------------------------------------------------
#define SMEM_HEAD 187392
__global__ void __launch_bounds__(256) dpp_fused_head(
    const float* __restrict__ x0, const float* __restrict__ rbf,
    const float* __restrict__ w_rbf, const float* __restrict__ w_ji,
    const float* __restrict__ b_ji, const float* __restrict__ w_kj,
    const float* __restrict__ b_kj, const float* __restrict__ w_down,
    const bf16_t* __restrict__ whead_bf,
    float* __restrict__ xji_out, float* __restrict__ xkjd_out) {
  extern __shared__ char smem[];
  bf16_t* wTji  = (bf16_t*)(smem);            // 128N x 128K bf16 (32KB)
  bf16_t* wTkj  = (bf16_t*)(smem + 32768);    // 32KB
  bf16_t* wTdn  = (bf16_t*)(smem + 65536);    // 64N x 128K bf16 (16KB)
  float*  bjiL  = (float*)(smem + 81920);     // 512B
  float*  bkjL  = (float*)(smem + 82432);     // 512B
  float*  rbfe  = (float*)(smem + 82944);     // 128x128 f32 (64KB)
  bf16_t* act   = (bf16_t*)(smem + 148480);   // 128x128 bf16 (32KB)
  float*  rbfL  = (float*)(smem + 181248);    // 128x6 f32
  float*  wrbfL = (float*)(smem + 184320);    // 6x128 f32

  int tid = threadIdx.x;
  long long rowbase = (long long)blockIdx.x * 128;

#if HAVE_TDM
  if (tid == 0) {
    tdm_load_weights(whead_bf, smem, 5);      // 5 x 16KB = wTji|wTkj|wTdn
    __builtin_amdgcn_s_wait_tensorcnt(0);
  }
#else
  for (int i = tid; i < EMB * EMB; i += 256) {
    int k = i >> 7, n = i & 127;
    wTji[n * 128 + k] = (bf16_t)w_ji[i];
    wTkj[n * 128 + k] = (bf16_t)w_kj[i];
  }
  for (int i = tid; i < EMB * INTD; i += 256) {
    int k = i >> 6, n = i & 63;
    wTdn[n * 128 + k] = (bf16_t)w_down[i];
  }
#endif
  for (int i = tid; i < 128; i += 256) { bjiL[i] = b_ji[i]; bkjL[i] = b_kj[i]; }
  for (int i = tid; i < 128 * NRBF; i += 256) rbfL[i] = rbf[rowbase * NRBF + i];
  for (int i = tid; i < NRBF * EMB; i += 256) wrbfL[i] = w_rbf[i];
  for (int i = tid; i < 128 * EMB; i += 256) act[i] = (bf16_t)x0[rowbase * EMB + i];
  __syncthreads();
  for (int i = tid; i < 128 * EMB; i += 256) {  // rbf_e tile
    int r = i >> 7, c = i & 127;
    float s = 0.f;
#pragma unroll
    for (int j = 0; j < NRBF; ++j) s += rbfL[r * NRBF + j] * wrbfL[j * EMB + c];
    rbfe[i] = s;
  }
  __syncthreads();

  int lane = tid & 31, wave = tid >> 5;
  int half = lane >> 4, nn = lane & 15;
  int mrow = wave * 16 + nn;

  bf16x16 a[4];
#pragma unroll
  for (int kt = 0; kt < 4; ++kt) a[kt] = ld_frag_A(act, EMB, mrow, kt, half);

  // x_ji (A fragments shared with x_kj GEMM)
#pragma unroll
  for (int nt = 0; nt < 8; ++nt) {
    bf16x16 bf[4];
#pragma unroll
    for (int kt = 0; kt < 4; ++kt)
      bf[kt] = ld_frag_B(wTji, EMB, nt * 16 + nn, kt, half);
    f32x8 acc;
    float b = bjiL[nt * 16 + nn];
#pragma unroll
    for (int e = 0; e < 8; ++e) acc[e] = b;
#pragma unroll
    for (int kt = 0; kt < 4; ++kt) acc = wmma_bf16(a[kt], bf[kt], acc);
#pragma unroll
    for (int e = 0; e < 8; ++e) {
      long long rg = rowbase + wave * 16 + e + 8 * half;
      xji_out[rg * EMB + nt * 16 + nn] = silu_f(acc[e]);
    }
  }
  // x_kj pre-down (overwrites the x0 tile -- A already in registers)
#pragma unroll
  for (int nt = 0; nt < 8; ++nt) {
    bf16x16 bf[4];
#pragma unroll
    for (int kt = 0; kt < 4; ++kt)
      bf[kt] = ld_frag_B(wTkj, EMB, nt * 16 + nn, kt, half);
    f32x8 acc;
    float b = bkjL[nt * 16 + nn];
#pragma unroll
    for (int e = 0; e < 8; ++e) acc[e] = b;
#pragma unroll
    for (int kt = 0; kt < 4; ++kt) acc = wmma_bf16(a[kt], bf[kt], acc);
#pragma unroll
    for (int e = 0; e < 8; ++e) {
      int mloc = wave * 16 + e + 8 * half;
      float v = silu_f(acc[e]) * rbfe[mloc * EMB + nt * 16 + nn];
      act[mloc * EMB + nt * 16 + nn] = (bf16_t)v;
    }
  }
  // down-projection, N = 64
#pragma unroll
  for (int kt = 0; kt < 4; ++kt) a[kt] = ld_frag_A(act, EMB, mrow, kt, half);
#pragma unroll
  for (int nt = 0; nt < 4; ++nt) {
    bf16x16 bf[4];
#pragma unroll
    for (int kt = 0; kt < 4; ++kt)
      bf[kt] = ld_frag_B(wTdn, EMB, nt * 16 + nn, kt, half);
    f32x8 acc;
#pragma unroll
    for (int e = 0; e < 8; ++e) acc[e] = 0.0f;
#pragma unroll
    for (int kt = 0; kt < 4; ++kt) acc = wmma_bf16(a[kt], bf[kt], acc);
#pragma unroll
    for (int e = 0; e < 8; ++e) {
      long long rg = rowbase + wave * 16 + e + 8 * half;
      xkjd_out[rg * INTD + nt * 16 + nn] = silu_f(acc[e]);
    }
  }
}

// ---------------------------------------------------------------------------
// Kernel 4: triplet gather * sbf_t, scatter-add via f32 atomics.
// ---------------------------------------------------------------------------
__global__ void __launch_bounds__(256) dpp_triplet_scatter(
    const float* __restrict__ sbf, const int* __restrict__ id_expand,
    const int* __restrict__ id_reduce, const float* __restrict__ xkjd,
    const float* __restrict__ w_sbf, float* __restrict__ seg) {
  __shared__ float wsbfL[NSBF * INTD];
  __shared__ float sbfL[4 * NSBF];
  int tid = threadIdx.x;
  long long base_t = (long long)blockIdx.x * 4;
  for (int i = tid; i < NSBF * INTD; i += 256) wsbfL[i] = w_sbf[i];
  for (int i = tid; i < 4 * NSBF; i += 256) sbfL[i] = sbf[base_t * NSBF + i];
  __syncthreads();

  int lt = tid >> 6;   // triplet within block
  int c  = tid & 63;   // channel
  long long t = base_t + lt;
  float s = 0.0f;
#pragma unroll
  for (int j = 0; j < NSBF; ++j) s += sbfL[lt * NSBF + j] * wsbfL[j * INTD + c];
  int kj = id_expand[t];
  int ji = id_reduce[t];
  float val = xkjd[(long long)kj * INTD + c] * s;
  unsafeAtomicAdd(&seg[(long long)ji * INTD + c], val);
}

// ---------------------------------------------------------------------------
// Kernel 5: fused tail. All weights (240KB bf16) arrive via ONE TDM
// descriptor (15 x 16KB rows into smem+0); 8-GEMM chain, barrier-free.
// ---------------------------------------------------------------------------
#define SMEM_TAIL 298496
__global__ void __launch_bounds__(256) dpp_fused_tail(
    const float* __restrict__ x0, const float* __restrict__ xji,
    const float* __restrict__ seg, const float* __restrict__ w_up,
    const float* __restrict__ wb1, const float* __restrict__ bb1,
    const float* __restrict__ wb2, const float* __restrict__ bb2,
    const float* __restrict__ w_fin, const float* __restrict__ b_fin,
    const float* __restrict__ wa1, const float* __restrict__ ba1,
    const float* __restrict__ wa2, const float* __restrict__ ba2,
    const bf16_t* __restrict__ wtail_bf, float* __restrict__ out) {
  extern __shared__ char smem[];
  bf16_t* wTup  = (bf16_t*)(smem);                    // 128N x 64K  (16KB)
  bf16_t* wT    = (bf16_t*)(smem + 16384);            // 7 x 128x128 (224KB)
  float*  biasL = (float*)(smem + 16384 + 229376);    // 7 x 128 f32
  bf16_t* act64 = (bf16_t*)(smem + 16384 + 229376 + 3584);          // 16KB
  bf16_t* act   = (bf16_t*)(smem + 16384 + 229376 + 3584 + 16384);  // 32KB

  const float* Bsrc[7] = {bb1, bb2, b_fin, ba1, ba2, ba1 + EMB, ba2 + EMB};

  int tid = threadIdx.x;
  long long rowbase = (long long)blockIdx.x * 128;

#if HAVE_TDM
  if (tid == 0) {
    tdm_load_weights(wtail_bf, smem, 15);     // 15 x 16KB = wTup + 7 weights
    __builtin_amdgcn_s_wait_tensorcnt(0);
  }
#else
  {
    const float* Wsrc[7] = {wb1, wb2, w_fin, wa1, wa2,
                            wa1 + EMB * EMB, wa2 + EMB * EMB};
#pragma unroll
    for (int w = 0; w < 7; ++w) {
      const float* g = Wsrc[w];
      bf16_t* dst = wT + w * EMB * EMB;
      for (int i = tid; i < EMB * EMB; i += 256) {
        int k = i >> 7, n = i & 127;
        dst[n * 128 + k] = (bf16_t)g[i];
      }
    }
    for (int i = tid; i < INTD * EMB; i += 256) {
      int k = i >> 7, n = i & 127;
      wTup[n * 64 + k] = (bf16_t)w_up[i];
    }
  }
#endif
#pragma unroll
  for (int w = 0; w < 7; ++w)
    for (int i = tid; i < 128; i += 256) biasL[w * 128 + i] = Bsrc[w][i];
  for (int i = tid; i < 128 * INTD; i += 256)
    act64[i] = (bf16_t)seg[rowbase * INTD + i];
  __syncthreads();

  int lane = tid & 31, wave = tid >> 5;
  int half = lane >> 4, nn = lane & 15;
  int mrow = wave * 16 + nn;

  float resid[8][8];  // f32 running residual, wave-private

  // Stage 1: x2 = silu(seg @ w_up) + x_ji   (K = 64)
  {
    bf16x16 a0 = ld_frag_A(act64, INTD, mrow, 0, half);
    bf16x16 a1 = ld_frag_A(act64, INTD, mrow, 1, half);
#pragma unroll
    for (int nt = 0; nt < 8; ++nt) {
      bf16x16 b0 = ld_frag_B(wTup, INTD, nt * 16 + nn, 0, half);
      bf16x16 b1 = ld_frag_B(wTup, INTD, nt * 16 + nn, 1, half);
      f32x8 acc;
#pragma unroll
      for (int e = 0; e < 8; ++e) acc[e] = 0.0f;
      acc = wmma_bf16(a0, b0, acc);
      acc = wmma_bf16(a1, b1, acc);
#pragma unroll
      for (int e = 0; e < 8; ++e) {
        int mloc = wave * 16 + e + 8 * half;
        long long rg = rowbase + mloc;
        float v = silu_f(acc[e]) + xji[rg * EMB + nt * 16 + nn];
        resid[nt][e] = v;
        act[mloc * EMB + nt * 16 + nn] = (bf16_t)v;
      }
    }
  }
  // Stage 2: t = silu(x2 @ wb1 + bb1)
  square_gemm(act, wT + 0 * EMB * EMB, biasL + 0 * 128, wave, half, nn,
              [&](int nt, f32x8 acc) {
#pragma unroll
                for (int e = 0; e < 8; ++e) {
                  int mloc = wave * 16 + e + 8 * half;
                  act[mloc * EMB + nt * 16 + nn] = (bf16_t)silu_f(acc[e]);
                }
              });
  // Stage 3: x2 = x2 + silu(t @ wb2 + bb2)
  square_gemm(act, wT + 1 * EMB * EMB, biasL + 1 * 128, wave, half, nn,
              [&](int nt, f32x8 acc) {
#pragma unroll
                for (int e = 0; e < 8; ++e) {
                  int mloc = wave * 16 + e + 8 * half;
                  float v = resid[nt][e] + silu_f(acc[e]);
                  resid[nt][e] = v;
                  act[mloc * EMB + nt * 16 + nn] = (bf16_t)v;
                }
              });
  // Stage 4: x = x0 + silu(x2 @ w_fin + b_fin)
  square_gemm(act, wT + 2 * EMB * EMB, biasL + 2 * 128, wave, half, nn,
              [&](int nt, f32x8 acc) {
#pragma unroll
                for (int e = 0; e < 8; ++e) {
                  int mloc = wave * 16 + e + 8 * half;
                  long long rg = rowbase + mloc;
                  float v = x0[rg * EMB + nt * 16 + nn] + silu_f(acc[e]);
                  resid[nt][e] = v;
                  act[mloc * EMB + nt * 16 + nn] = (bf16_t)v;
                }
              });
  // Stage 5/6: residual layer 0
  square_gemm(act, wT + 3 * EMB * EMB, biasL + 3 * 128, wave, half, nn,
              [&](int nt, f32x8 acc) {
#pragma unroll
                for (int e = 0; e < 8; ++e) {
                  int mloc = wave * 16 + e + 8 * half;
                  act[mloc * EMB + nt * 16 + nn] = (bf16_t)silu_f(acc[e]);
                }
              });
  square_gemm(act, wT + 4 * EMB * EMB, biasL + 4 * 128, wave, half, nn,
              [&](int nt, f32x8 acc) {
#pragma unroll
                for (int e = 0; e < 8; ++e) {
                  int mloc = wave * 16 + e + 8 * half;
                  float v = resid[nt][e] + silu_f(acc[e]);
                  resid[nt][e] = v;
                  act[mloc * EMB + nt * 16 + nn] = (bf16_t)v;
                }
              });
  // Stage 7/8: residual layer 1 (+ final store)
  square_gemm(act, wT + 5 * EMB * EMB, biasL + 5 * 128, wave, half, nn,
              [&](int nt, f32x8 acc) {
#pragma unroll
                for (int e = 0; e < 8; ++e) {
                  int mloc = wave * 16 + e + 8 * half;
                  act[mloc * EMB + nt * 16 + nn] = (bf16_t)silu_f(acc[e]);
                }
              });
  square_gemm(act, wT + 6 * EMB * EMB, biasL + 6 * 128, wave, half, nn,
              [&](int nt, f32x8 acc) {
#pragma unroll
                for (int e = 0; e < 8; ++e) {
                  int mloc = wave * 16 + e + 8 * half;
                  long long rg = rowbase + mloc;
                  float v = resid[nt][e] + silu_f(acc[e]);
                  out[rg * EMB + nt * 16 + nn] = v;
                }
              });
}

// ---------------------------------------------------------------------------
extern "C" void kernel_launch(void* const* d_in, const int* in_sizes, int n_in,
                              void* d_out, int out_size, void* d_ws, size_t ws_size,
                              hipStream_t stream) {
  const float* x0     = (const float*)d_in[0];
  const float* rbf    = (const float*)d_in[1];
  const float* sbf    = (const float*)d_in[2];
  const int* id_expand = (const int*)d_in[3];
  const int* id_reduce = (const int*)d_in[4];
  // d_in[5] = R (unused)
  const float* w_rbf1 = (const float*)d_in[6];
  const float* w_rbf2 = (const float*)d_in[7];
  const float* w_sbf1 = (const float*)d_in[8];
  const float* w_sbf2 = (const float*)d_in[9];
  const float* w_ji   = (const float*)d_in[10];
  const float* b_ji   = (const float*)d_in[11];
  const float* w_kj   = (const float*)d_in[12];
  const float* b_kj   = (const float*)d_in[13];
  const float* w_down = (const float*)d_in[14];
  const float* w_up   = (const float*)d_in[15];
  const float* wb1    = (const float*)d_in[16];
  const float* bb1    = (const float*)d_in[17];
  const float* wb2    = (const float*)d_in[18];
  const float* bb2    = (const float*)d_in[19];
  const float* w_fin  = (const float*)d_in[20];
  const float* b_fin  = (const float*)d_in[21];
  const float* wa1    = (const float*)d_in[22];
  const float* ba1    = (const float*)d_in[23];
  const float* wa2    = (const float*)d_in[24];
  const float* ba2    = (const float*)d_in[25];
  float* out = (float*)d_out;

  // workspace layout (floats), bf16 weight images appended at the end
  float* ws    = (float*)d_ws;
  float* w_rbf = ws;                                // 6*128
  float* w_sbf = ws + 1024;                         // 42*64
  float* xji   = ws + 4096;                         // E*128
  float* xkjd  = xji + (size_t)E_EDGES * EMB;       // E*64
  float* seg   = xkjd + (size_t)E_EDGES * INTD;     // E*64
  bf16_t* wtail_bf = (bf16_t*)(seg + (size_t)E_EDGES * INTD);  // 122880 bf16
  bf16_t* whead_bf = wtail_bf + 122880;                        // 40960 bf16

  hipFuncSetAttribute(reinterpret_cast<const void*>(dpp_fused_head),
                      hipFuncAttributeMaxDynamicSharedMemorySize, SMEM_HEAD);
  hipFuncSetAttribute(reinterpret_cast<const void*>(dpp_fused_tail),
                      hipFuncAttributeMaxDynamicSharedMemorySize, SMEM_TAIL);

  dpp_precompute_basis<<<1, 256, 0, stream>>>(w_rbf1, w_rbf2, w_sbf1, w_sbf2,
                                              w_rbf, w_sbf);
  dpp_zero_f32<<<(E_EDGES * INTD) / 256, 256, 0, stream>>>(
      seg, (long long)E_EDGES * INTD);
  dpp_prep_weights<<<480, 256, 0, stream>>>(w_up, wb1, wb2, w_fin, wa1, wa2,
                                            w_ji, w_kj, w_down,
                                            wtail_bf, whead_bf);
  dpp_fused_head<<<E_EDGES / 128, 256, SMEM_HEAD, stream>>>(
      x0, rbf, w_rbf, w_ji, b_ji, w_kj, b_kj, w_down, whead_bf, xji, xkjd);
  dpp_triplet_scatter<<<T_TRIP / 4, 256, 0, stream>>>(
      sbf, id_expand, id_reduce, xkjd, w_sbf, seg);
  dpp_fused_tail<<<E_EDGES / 128, 256, SMEM_TAIL, stream>>>(
      x0, xji, seg, w_up, wb1, bb1, wb2, bb2, w_fin, b_fin,
      wa1, ba1, wa2, ba2, wtail_bf, out);
}
